// MultiHeadSelfAttention_30236569763899
// MI455X (gfx1250) — matrix operand order repaired and checked by reading, not compiled, not run on hardware
//
#include <hip/hip_runtime.h>

// MI455X / gfx1250, wave32. bf16 WMMA path (v_wmma_f32_16x16x32_bf16)
// + async global->LDS staging (ASYNCcnt) when the toolchain exposes it.

typedef __attribute__((ext_vector_type(16))) __bf16 v16bf;
typedef __attribute__((ext_vector_type(8)))  __bf16 v8bf;
typedef __attribute__((ext_vector_type(8)))  float  v8f;
typedef __attribute__((ext_vector_type(4)))  int    v4i;

#if defined(__AMDGCN__) && __has_builtin(__builtin_amdgcn_global_load_async_to_lds_b128) && \
    __has_builtin(__builtin_amdgcn_s_wait_asynccnt)
#define USE_ASYNC 1
#else
#define USE_ASYNC 0
#endif

#if USE_ASYNC
#define AS1 __attribute__((address_space(1)))
#define AS3 __attribute__((address_space(3)))
typedef AS1 v4i* gptr_b128;
typedef AS3 v4i* lptr_b128;
static __device__ inline gptr_b128 to_global_b128(const void* p) {
    return (gptr_b128)(uintptr_t)p;              // flat global addr == AS1 addr
}
static __device__ inline lptr_b128 to_lds_b128(void* p) {
    return (lptr_b128)(uint32_t)(uintptr_t)p;    // low 32 bits = LDS offset
}
#define ASYNC_WAIT(n) __builtin_amdgcn_s_wait_asynccnt(n)
#else
#define ASYNC_WAIT(n)
#endif

static __device__ inline v16bf bf_concat(v8bf lo, v8bf hi) {
    return __builtin_shufflevector(lo, hi, 0,1,2,3,4,5,6,7,8,9,10,11,12,13,14,15);
}

static __device__ inline v8f wmma_bf16(v16bf a, v16bf b, v8f c) {
    return __builtin_amdgcn_wmma_f32_16x16x32_bf16(false, a, false, b, (short)0, c, false, false);
}

// ---------------------------------------------------------------------------
// Pre-pass A: Qbf[i] = bf16(Q[i]).  4096x1024 elements, float4-vectorized.
// ---------------------------------------------------------------------------
__global__ __launch_bounds__(256)
void cvt_q_kernel(const float* __restrict__ Q, __bf16* __restrict__ Qbf)
{
    const size_t i = ((size_t)blockIdx.x * 256 + threadIdx.x) * 4;
    const float4 v = *(const float4*)(Q + i);
    __bf16* d = Qbf + i;
    d[0] = (__bf16)v.x; d[1] = (__bf16)v.y;
    d[2] = (__bf16)v.z; d[3] = (__bf16)v.w;
}

// ---------------------------------------------------------------------------
// Pre-pass B: WT[z][n][k] = bf16(W_z[k][n])  (32x32 LDS tile transpose).
// ---------------------------------------------------------------------------
__global__ __launch_bounds__(256)
void wtrans_kernel(const float* __restrict__ Wq, const float* __restrict__ Wk,
                   const float* __restrict__ Wv, __bf16* __restrict__ WT)
{
    const int z = blockIdx.z;
    const float* __restrict__ W = (z == 0) ? Wq : ((z == 1) ? Wk : Wv);
    __bf16* __restrict__ dst = WT + (size_t)z * 1024 * 1024;

    const int n0 = blockIdx.x * 32;
    const int k0 = blockIdx.y * 32;
    __shared__ float tile[32][33];

    const int tr = threadIdx.x >> 3;         // 0..31
    const int tc = (threadIdx.x & 7) * 4;    // 0,4,...,28

    const float4 v = *(const float4*)(W + (size_t)(k0 + tr) * 1024 + n0 + tc);
    tile[tr][tc + 0] = v.x; tile[tr][tc + 1] = v.y;
    tile[tr][tc + 2] = v.z; tile[tr][tc + 3] = v.w;
    __syncthreads();

    __bf16* o = dst + (size_t)(n0 + tr) * 1024 + k0 + tc;
    o[0] = (__bf16)tile[tc + 0][tr];
    o[1] = (__bf16)tile[tc + 1][tr];
    o[2] = (__bf16)tile[tc + 2][tr];
    o[3] = (__bf16)tile[tc + 3][tr];
}

// ---------------------------------------------------------------------------
// Kernel 1: fused QKV projection.  Y = Qbf @ W + b, all-bf16 tiles, double-
// buffered async global->LDS staging (16B chunks), 8 waves x (16x32) WMMA.
// Outputs (bf16): q,k -> [B,H,S,DK];  v -> transposed [B,H,DK,S].
// ---------------------------------------------------------------------------
__global__ __launch_bounds__(256)
void qkv_proj_kernel(const __bf16* __restrict__ Qbf,
                     const __bf16* __restrict__ WT,
                     const float* __restrict__ bq, const float* __restrict__ bk,
                     const float* __restrict__ bv,
                     __bf16* __restrict__ qh, __bf16* __restrict__ kh,
                     __bf16* __restrict__ vT)
{
    const int z = blockIdx.z;                       // 0:q 1:k 2:v (uniform)
    const float* __restrict__ bias = (z == 0) ? bq : ((z == 1) ? bk : bv);
    const __bf16* __restrict__ Wz = WT + (size_t)z * 1024 * 1024;

    const int mBase = blockIdx.x * 64;
    const int nBase = blockIdx.y * 64;
    const int tid  = threadIdx.x;
    const int w    = tid >> 5;
    const int lane = tid & 31;
    const int wm = w & 3, wn = w >> 2;
    const int half = lane >> 4, r16 = lane & 15;

    // Row pitch 40 bf16 (80 B): 16B-aligned rows, conflict-free frag reads.
    __shared__ __bf16 Asm[2][64 * 40];   // X tile [m][k]
    __shared__ __bf16 Bsm[2][64 * 40];   // W tile [n][k] (pre-transposed)

    // Cooperative copy geometry: one 16B chunk per thread per tile.
    const int crow = tid >> 2;           // 0..63
    const int ckk  = (tid & 3) * 8;      // 0,8,16,24
    const __bf16* __restrict__ gA = Qbf + (size_t)(mBase + crow) * 1024 + ckk;
    const __bf16* __restrict__ gB = Wz  + (size_t)(nBase + crow) * 1024 + ckk;
    const int ldsOff = crow * 40 + ckk;

    auto issue_tile = [&](int buf, int k0) {
#if USE_ASYNC
        __builtin_amdgcn_global_load_async_to_lds_b128(
            to_global_b128(gA + k0), to_lds_b128(&Asm[buf][ldsOff]), 0, 0);
        __builtin_amdgcn_global_load_async_to_lds_b128(
            to_global_b128(gB + k0), to_lds_b128(&Bsm[buf][ldsOff]), 0, 0);
#else
        *(v8bf*)&Asm[buf][ldsOff] = *(const v8bf*)(gA + k0);
        *(v8bf*)&Bsm[buf][ldsOff] = *(const v8bf*)(gB + k0);
#endif
    };

    v8f acc0 = {}; v8f acc1 = {};

    auto compute_tile = [&](int buf) {
        // A frag (16x32): lanes 0-15 -> K 0-15; lanes 16-31 -> K 16-31
        const __bf16* arow = &Asm[buf][(wm * 16 + r16) * 40];
        v16bf afrag = bf_concat(*(const v8bf*)&arow[half * 16],
                                *(const v8bf*)&arow[half * 16 + 8]);
        // B frags (32x16): lanes 0-15 -> K 0-15; lanes 16-31 -> K 16-31
        const __bf16* b0row = &Bsm[buf][(wn * 32 +  0 + r16) * 40];
        const __bf16* b1row = &Bsm[buf][(wn * 32 + 16 + r16) * 40];
        v16bf bf0 = bf_concat(*(const v8bf*)&b0row[half * 16],
                              *(const v8bf*)&b0row[half * 16 + 8]);
        v16bf bf1 = bf_concat(*(const v8bf*)&b1row[half * 16],
                              *(const v8bf*)&b1row[half * 16 + 8]);
        acc0 = wmma_bf16(afrag, bf0, acc0);
        acc1 = wmma_bf16(afrag, bf1, acc1);
    };

    // Double-buffered K loop: 32 tiles of K=32.
    issue_tile(0, 0);
    int buf = 0;
    for (int k0 = 0; k0 < 1024 - 32; k0 += 32) {
        issue_tile(buf ^ 1, k0 + 32);
        ASYNC_WAIT(2);                 // my copies for `buf` have landed
        __syncthreads();               // everyone's copies for `buf` landed
        compute_tile(buf);
        __syncthreads();               // all reads of `buf` done before reuse
        buf ^= 1;
    }
    ASYNC_WAIT(0);
    __syncthreads();
    compute_tile(buf);

    // Epilogue: C/D row = r + 8*half, col = r16 within each 16x16 tile.
    const int m0 = mBase + wm * 16 + half * 8;
    #pragma unroll
    for (int j = 0; j < 2; ++j) {
        const int n  = nBase + wn * 32 + j * 16 + r16;
        const float bv_ = bias[n];
        const int hh = n >> 6, dk = n & 63;
        v8f acc = j ? acc1 : acc0;
        #pragma unroll
        for (int r = 0; r < 8; ++r) {
            const int m  = m0 + r;
            const int bb = m >> 11;
            const int ss = m & 2047;
            const float val = acc[r] + bv_;
            const size_t bhh = (size_t)bb * 16 + hh;
            if (z == 0)      qh[(bhh * 2048 + ss) * 64 + dk] = (__bf16)val;
            else if (z == 1) kh[(bhh * 2048 + ss) * 64 + dk] = (__bf16)val;
            else             vT[(bhh * 64 + dk) * 2048 + ss] = (__bf16)val;
        }
    }
}

// ---------------------------------------------------------------------------
// Kernel 2: streaming attention, raw exp (faithful), multiplicative key mask.
// Block: 8 waves x 16 q-rows; keys in groups of 32 up to ceil(len/32).
// P transpose goes through a wave-private LDS tile; same-wave LDS ops are
// architecturally in-order (DScnt), so no block barriers in the k-loop.
// ---------------------------------------------------------------------------
__global__ __launch_bounds__(256)
void attn_kernel(const __bf16* __restrict__ qh, const __bf16* __restrict__ kh,
                 const __bf16* __restrict__ vT, const int* __restrict__ length,
                 float* __restrict__ out)
{
    const int b = blockIdx.z, h = blockIdx.y;
    const int qbase = blockIdx.x * 128;
    const int tid  = threadIdx.x;
    const int w    = tid >> 5;
    const int lane = tid & 31;
    const int half = lane >> 4, r16 = lane & 15;

    const int len  = length[b];
    const int ngrp = (len + 31) >> 5;

    __shared__ float Psm[8][16][36];   // per-wave 16x32 tile, padded pitch

    const size_t bh = (size_t)b * 16 + h;
    const __bf16* __restrict__ qrow = qh + (bh * 2048 + (qbase + w * 16 + r16)) * 64;
    const __bf16* __restrict__ kpan = kh + bh * 2048 * 64;
    const __bf16* __restrict__ vpan = vT + bh * 64 * 2048;

    v16bf q0 = bf_concat(*(const v8bf*)&qrow[half * 16],
                         *(const v8bf*)&qrow[half * 16 + 8]);
    v16bf q1 = bf_concat(*(const v8bf*)&qrow[32 + half * 16],
                         *(const v8bf*)&qrow[32 + half * 16 + 8]);

    v8f c0 = {}, c1 = {}, c2 = {}, c3 = {};
    float dsum[8] = {};

    for (int g = 0; g < ngrp; ++g) {
        #pragma unroll
        for (int t = 0; t < 2; ++t) {
            const int key = g * 32 + t * 16 + r16;
            const __bf16* krow = kpan + (size_t)key * 64;
            v16bf kb0 = bf_concat(*(const v8bf*)&krow[half * 16],
                                  *(const v8bf*)&krow[half * 16 + 8]);
            v16bf kb1 = bf_concat(*(const v8bf*)&krow[32 + half * 16],
                                  *(const v8bf*)&krow[32 + half * 16 + 8]);
            v8f s = {};
            s = wmma_bf16(q0, kb0, s);
            s = wmma_bf16(q1, kb1, s);
            #pragma unroll
            for (int r = 0; r < 8; ++r) {
                const float p = (key < len) ? __expf(s[r] * 0.125f) : 0.0f;
                dsum[r] += p;                       // row r + 8*half
                Psm[w][r + half * 8][t * 16 + r16] = p;
            }
        }
        // Same-wave LDS store->load: in-order at the LDS unit; no barrier.
        v16bf pf;
        #pragma unroll
        for (int e = 0; e < 8; ++e)
            pf[e] = (__bf16)Psm[w][r16][half * 16 + e];
        #pragma unroll
        for (int e = 0; e < 8; ++e)
            pf[8 + e] = (__bf16)Psm[w][r16][half * 16 + 8 + e];

        #pragma unroll
        for (int j = 0; j < 4; ++j) {
            const __bf16* vrow = vpan + (size_t)(j * 16 + r16) * 2048 + g * 32 + half * 16;
            v16bf vb = bf_concat(*(const v8bf*)&vrow[0], *(const v8bf*)&vrow[8]);
            if      (j == 0) c0 = wmma_bf16(pf, vb, c0);
            else if (j == 1) c1 = wmma_bf16(pf, vb, c1);
            else if (j == 2) c2 = wmma_bf16(pf, vb, c2);
            else             c3 = wmma_bf16(pf, vb, c3);
        }
    }

    #pragma unroll
    for (int r = 0; r < 8; ++r) {
        float d = dsum[r];
        d += __shfl_xor(d, 1, 32);
        d += __shfl_xor(d, 2, 32);
        d += __shfl_xor(d, 4, 32);
        d += __shfl_xor(d, 8, 32);
        dsum[r] = 1.0f / (d + 1e-8f);
    }

    #pragma unroll
    for (int j = 0; j < 4; ++j) {
        v8f c = (j == 0) ? c0 : ((j == 1) ? c1 : ((j == 2) ? c2 : c3));
        const int dcol = h * 64 + j * 16 + r16;
        #pragma unroll
        for (int r = 0; r < 8; ++r) {
            const int s = qbase + w * 16 + r + half * 8;
            out[((size_t)b * 2048 + s) * 1024 + dcol] = c[r] * dsum[r];
        }
    }
}

// ---------------------------------------------------------------------------
extern "C" void kernel_launch(void* const* d_in, const int* in_sizes, int n_in,
                              void* d_out, int out_size, void* d_ws, size_t ws_size,
                              hipStream_t stream)
{
    (void)in_sizes; (void)n_in; (void)out_size; (void)ws_size;

    const float* Q      = (const float*)d_in[0];
    const int*   length = (const int*)  d_in[1];
    const float* Wq     = (const float*)d_in[2];
    const float* bq     = (const float*)d_in[3];
    const float* Wk     = (const float*)d_in[4];
    const float* bk     = (const float*)d_in[5];
    const float* Wv     = (const float*)d_in[6];
    const float* bv     = (const float*)d_in[7];
    float* out = (float*)d_out;

    // Workspace layout (bf16 elements):
    //   qh/kh/vT : 3 x 4,194,304   Qbf : 4,194,304   WT : 3,145,728  (~38 MB)
    __bf16* qh  = (__bf16*)d_ws;
    __bf16* khp = qh  + (size_t)4194304;
    __bf16* vT  = khp + (size_t)4194304;
    __bf16* Qbf = vT  + (size_t)4194304;
    __bf16* WT  = Qbf + (size_t)4194304;

    dim3 blk(256);
    cvt_q_kernel<<<dim3(4096), blk, 0, stream>>>(Q, Qbf);
    wtrans_kernel<<<dim3(32, 32, 3), blk, 0, stream>>>(Wq, Wk, Wv, WT);
    qkv_proj_kernel<<<dim3(64, 16, 3), blk, 0, stream>>>(Qbf, WT, bq, bk, bv,
                                                         qh, khp, vT);
    attn_kernel<<<dim3(16, 16, 2), blk, 0, stream>>>(qh, khp, vT, length, out);
}
